// GammaFragmentModel_87196426043468
// MI455X (gfx1250) — compile-verified, matrix operand order checked by reading, not run on hardware
//
#include <hip/hip_runtime.h>
#include <hip/hip_bf16.h>
#include <cstdint>
#include <cstddef>

// ---------------------------------------------------------------------------
// GNN forward (3x message passing: BN -> edge MLP -> GATConv) for gfx1250.
// Bandwidth-bound workload: all GEMMs are K<=32, width 32 -> mapped to
// v_wmma_f32_16x16x32_bf16 with BN folded into weights on-device.
// Weights ride in the A operand (M = out channel), rows in the B operand
// (N = row-in-tile) so each lane's D fragment is 8 consecutive channels of
// one row -> single b128 store, and row loads are two contiguous 16B runs.
// All hot-loop index math is 32-bit (E = 1.6M, max element count 51.2M).
// ---------------------------------------------------------------------------

typedef __attribute__((ext_vector_type(16))) __bf16        v16bf;
typedef __attribute__((ext_vector_type(8)))  float         v8f;
typedef __attribute__((ext_vector_type(4)))  float         v4f;
typedef __attribute__((ext_vector_type(8)))  unsigned int  v8u;

#define EPS_BN 1e-5f

// lrelu(x, s) == max(x, s*x) for 0 < s <= 1 (s=1 -> identity): 1 VALU op/elem.
__device__ __forceinline__ float lrelu(float x, float s){ return __builtin_fmaxf(x, s * x); }
__device__ __forceinline__ unsigned short f2bf(float f){ __bf16 h = (__bf16)f; return __builtin_bit_cast(unsigned short, h); }
__device__ __forceinline__ float bf2f(unsigned short u){ __bf16 h = __builtin_bit_cast(__bf16, u); return (float)h; }

__device__ __forceinline__ v8f wmma_bf16(v16bf a, v16bf b, v8f c){
  // D = A x B + C ; A: 16x32 bf16 (M=out-channel), B: 32x16 bf16 (N=row), C/D f32
  return __builtin_amdgcn_wmma_f32_16x16x32_bf16(false, a, false, b, (short)0, c, false, false);
}

// 8 consecutive floats (two aligned float4s) -> v8f, register-only.
__device__ __forceinline__ v8f load8f(const float* __restrict__ p){
  v4f a = *(const v4f*)p;
  v4f b = *(const v4f*)(p + 4);
  return __builtin_shufflevector(a, b, 0, 1, 2, 3, 4, 5, 6, 7);
}

__device__ __forceinline__ void store8f(float* p, v8f d){
  *(v4f*)p       = __builtin_shufflevector(d, d, 0, 1, 2, 3);
  *(v4f*)(p + 4) = __builtin_shufflevector(d, d, 4, 5, 6, 7);
}

__device__ __forceinline__ uint4 pack_bf16x8(v8f x){
  uint4 u;
  u.x = (unsigned)f2bf(x[0]) | ((unsigned)f2bf(x[1]) << 16);
  u.y = (unsigned)f2bf(x[2]) | ((unsigned)f2bf(x[3]) << 16);
  u.z = (unsigned)f2bf(x[4]) | ((unsigned)f2bf(x[5]) << 16);
  u.w = (unsigned)f2bf(x[6]) | ((unsigned)f2bf(x[7]) << 16);
  return u;
}

__device__ __forceinline__ v8f lrelu8(v8f x, float s){
  v8f r;
#pragma unroll
  for (int i = 0; i < 8; i++) r[i] = __builtin_fmaxf(x[i], s * x[i]);
  return r;
}

// A fragment: A[m][k] = W[k][jBase+m]  (W row-major [Krows<=32, 32] f32)
// 16-bit A layout (ISA 7.12.2): lane m=l&15, half=l>>4, elem e -> k=(e&7)+8*half+16*(e>>3)
__device__ __forceinline__ v16bf load_wfragA(const float* __restrict__ W, int jBase, int lane, int Krows){
  int j = jBase + (lane & 15);
  int half = (lane >> 4) & 1;
  v16bf a;
#pragma unroll
  for (int e = 0; e < 16; e++){
    int k = (e & 7) + 8 * half + 16 * (e >> 3);
    float v = (k < Krows) ? W[k * 32 + j] : 0.f;
    a[e] = (__bf16)v;
  }
  return a;
}

// B fragment from a bf16 row: elems 0..7 = run at byte 16*half, 8..15 = run at
// byte 32+16*half (zero when CE==16). Register-only via v8u bit_cast.
template <int CE>
__device__ __forceinline__ v16bf load_bfrag_bf16(const unsigned short* __restrict__ rows, int row, int half){
  const uint4* p = (const uint4*)(rows + (size_t)((unsigned)row) * (unsigned)CE);
  uint4 r0 = p[half];
  uint4 r1 = (CE == 32) ? p[2 + half] : make_uint4(0u, 0u, 0u, 0u);
  v8u u;
  u[0] = r0.x; u[1] = r0.y; u[2] = r0.z; u[3] = r0.w;
  u[4] = r1.x; u[5] = r1.y; u[6] = r1.z; u[7] = r1.w;
  return __builtin_bit_cast(v16bf, u);
}

__device__ __forceinline__ void atomicMaxF(float* addr, float val){
  if (val >= 0.f) atomicMax((int*)addr, __float_as_int(val));
  else            atomicMin((unsigned int*)addr, __float_as_uint(val));
}

// ------------------------------ utility kernels ----------------------------

__global__ void k_zero(unsigned int* p, int n){
  int i = blockIdx.x * blockDim.x + threadIdx.x;
  if (i < n) p[i] = 0u;
}

__global__ void k_cvt_bf16(const float* __restrict__ in, unsigned short* __restrict__ out, int n){
  int i = blockIdx.x * blockDim.x + threadIdx.x;
  if (i < n) out[i] = f2bf(in[i]);
}

__global__ void k_deg(const int* __restrict__ s, const int* __restrict__ d, int E,
                      int* __restrict__ degs, int* __restrict__ degd){
  int j = blockIdx.x * blockDim.x + threadIdx.x;
  if (j < E){ atomicAdd(&degs[s[j]], 1); atomicAdd(&degd[d[j]], 1); }
}

__global__ void k_layer_init(float* __restrict__ small, float* __restrict__ amax,
                             float* __restrict__ den, float* __restrict__ xout, int N){
  int i = blockIdx.x * blockDim.x + threadIdx.x;
  if (i < 320) small[i] = 0.f;
  if (i < N){ amax[i] = -3.402823466e38f; den[i] = 0.f; }
  if (i < N * 32) xout[i] = 0.f;
}

// per-column sum & sumsq over row-major [n/C, C] (C power of two <= 32)
__global__ void k_colstats(const float* __restrict__ x, int n, int C,
                           float* __restrict__ sum, float* __restrict__ sq){
  __shared__ float ls[64];
  int t = threadIdx.x;
  if (t < 2 * C) ls[t] = 0.f;
  __syncthreads();
  int i = blockIdx.x * blockDim.x + t;
  int stride = gridDim.x * blockDim.x;
  int mask = C - 1;
  for (; i < n; i += stride){
    float v = x[i]; int c = i & mask;
    atomicAdd(&ls[c], v); atomicAdd(&ls[C + c], v * v);
  }
  __syncthreads();
  if (t < C){ atomicAdd(&sum[t], ls[t]); atomicAdd(&sq[t], ls[C + t]); }
}

// degree-weighted stats of BN'd node features (bn1 stats of gathered columns)
__global__ void k_bn1x_stats(const float* __restrict__ x, int N, int C, int csh,
                             const float* __restrict__ sc, const float* __restrict__ sh,
                             const int* __restrict__ degs, const int* __restrict__ degd,
                             float* s1s, float* q1s, float* s1d, float* q1d){
  __shared__ float ls[128];
  int t = threadIdx.x;
  if (t < 128) ls[t] = 0.f;
  __syncthreads();
  int n = N * C;
  int i = blockIdx.x * blockDim.x + t;
  int stride = gridDim.x * blockDim.x;
  int mask = C - 1;
  for (; i < n; i += stride){
    int c = i & mask; int nd = i >> csh;
    float xb = x[i] * sc[c] + sh[c];
    float ws_ = (float)degs[nd], wd_ = (float)degd[nd];
    atomicAdd(&ls[c],      ws_ * xb);
    atomicAdd(&ls[32 + c], ws_ * xb * xb);
    atomicAdd(&ls[64 + c], wd_ * xb);
    atomicAdd(&ls[96 + c], wd_ * xb * xb);
  }
  __syncthreads();
  if (t < 32){
    atomicAdd(&s1s[t], ls[t]);      atomicAdd(&q1s[t], ls[32 + t]);
    atomicAdd(&s1d[t], ls[64 + t]); atomicAdd(&q1d[t], ls[96 + t]);
  }
}

// node BN affine: scale = g*rsqrt(var+eps), shift = b - mean*scale
__global__ void k_fold_node(const float* sum, const float* sq, const float* g, const float* b,
                            long n, int C, float* sc, float* sh){
  int t = threadIdx.x;
  if (t < C){
    double m = (double)sum[t] / (double)n;
    double v = (double)sq[t] / (double)n - m * m;
    float s = g[t] * rsqrtf((float)v + EPS_BN);
    sc[t] = s; sh[t] = b[t] - (float)m * s;
  }
}

// Fold concat-BN (bn1) into W1 split by segment {src,dst,e}; also b1_eff.
// Zeros s1e/q1e afterwards so the next stage can re-accumulate.
__global__ void k_fold1(const float* __restrict__ w1, const float* __restrict__ g, const float* __restrict__ bb,
                        const float* __restrict__ blin,
                        const float* s1s, const float* q1s, const float* s1d, const float* q1d,
                        float* s1e, float* q1e,
                        int ni, int Ke, long E,
                        float* __restrict__ W1s, float* __restrict__ W1d, float* __restrict__ W1e,
                        float* __restrict__ beff){
  __shared__ float sc[96], sh[96];
  int t = threadIdx.x;
  if (t < 96){
    int seg = t >> 5, c = t & 31;
    int lim = (seg < 2) ? ni : Ke;
    float scv = 0.f, shv = 0.f;
    if (c < lim){
      const float* S = (seg == 0) ? s1s : ((seg == 1) ? s1d : s1e);
      const float* Q = (seg == 0) ? q1s : ((seg == 1) ? q1d : q1e);
      double m = (double)S[c] / (double)E;
      double v = (double)Q[c] / (double)E - m * m;
      int gi = (seg == 0) ? c : ((seg == 1) ? ni + c : 2 * ni + c);
      float s = g[gi] * rsqrtf((float)v + EPS_BN);
      scv = s; shv = bb[gi] - (float)m * s;
    }
    sc[t] = scv; sh[t] = shv;
  }
  __syncthreads();
  for (int i = t; i < 3 * 32 * 32; i += blockDim.x){
    int seg = i >> 10, k = (i >> 5) & 31, j = i & 31;
    int lim = (seg < 2) ? ni : Ke;
    float wv = 0.f;
    if (k < lim){
      int rk = (seg == 0) ? k : ((seg == 1) ? ni + k : 2 * ni + k);
      wv = sc[seg * 32 + k] * w1[rk * 32 + j];
    }
    float* W = (seg == 0) ? W1s : ((seg == 1) ? W1d : W1e);
    W[k * 32 + j] = wv;
  }
  __syncthreads();
  if (t < 32){
    float acc = blin[t];
    for (int k = 0; k < ni; k++) acc += sh[k]      * w1[k * 32 + t];
    for (int k = 0; k < ni; k++) acc += sh[32 + k] * w1[(ni + k) * 32 + t];
    for (int k = 0; k < Ke; k++) acc += sh[64 + k] * w1[(2 * ni + k) * 32 + t];
    beff[t] = acc;
  }
  __syncthreads();
  if (t < 32){ s1e[t] = 0.f; q1e[t] = 0.f; }
}

// Fold a 32-channel BN into a [32,32] linear.
__global__ void k_fold32(const float* __restrict__ w, const float* __restrict__ bng, const float* __restrict__ bnb,
                         const float* __restrict__ blin, const float* __restrict__ s, const float* __restrict__ q,
                         long E, float* __restrict__ Wf, float* __restrict__ beff){
  __shared__ float sc[32], sh[32];
  int t = threadIdx.x;
  if (t < 32){
    double m = (double)s[t] / (double)E;
    double v = (double)q[t] / (double)E - m * m;
    float scv = bng[t] * rsqrtf((float)v + EPS_BN);
    sc[t] = scv; sh[t] = bnb[t] - (float)m * scv;
  }
  __syncthreads();
  for (int i = t; i < 1024; i += blockDim.x){
    int k = i >> 5;
    Wf[i] = sc[k] * w[i];
  }
  if (t < 32){
    float acc = blin[t];
    for (int k = 0; k < 32; k++) acc += sh[k] * w[k * 32 + t];
    beff[t] = acc;
  }
}

// ------------------------------ WMMA kernels -------------------------------

// Per-node GEMMs: ps = xbn@W1s, pd = xbn@W1d, xl = xbn@gat_w (BN on the fly).
template <int C>
__global__ void k_node_mm_t(const float* __restrict__ x, int N,
                            const float* __restrict__ scN, const float* __restrict__ shN,
                            const float* __restrict__ W1s, const float* __restrict__ W1d,
                            const float* __restrict__ gatw,
                            float* __restrict__ ps, float* __restrict__ pd, float* __restrict__ xl){
  const int lane = threadIdx.x & 31;
  const int half = lane >> 4;
  const int wave = (blockIdx.x * blockDim.x + threadIdx.x) >> 5;
  const int nw = (gridDim.x * blockDim.x) >> 5;

  v8f sc0 = load8f(scN + 8 * half), sh0 = load8f(shN + 8 * half);
  v8f sc1 = {}, sh1 = {};
  if (C == 32){ sc1 = load8f(scN + 16 + 8 * half); sh1 = load8f(shN + 16 + 8 * half); }

  v16bf aS0 = load_wfragA(W1s, 0, lane, 32),  aS1 = load_wfragA(W1s, 16, lane, 32);
  v16bf aD0 = load_wfragA(W1d, 0, lane, 32),  aD1 = load_wfragA(W1d, 16, lane, 32);
  v16bf aG0 = load_wfragA(gatw, 0, lane, C),  aG1 = load_wfragA(gatw, 16, lane, C);

  const int ntiles = (N + 15) >> 4;
  for (int t = wave; t < ntiles; t += nw){
    int node = t * 16 + (lane & 15);
    int nc = node < N ? node : N - 1;
    const float* xr = x + (size_t)((unsigned)nc) * (unsigned)C;
    v8f xa = load8f(xr + 8 * half) * sc0 + sh0;
    v16bf b;
#pragma unroll
    for (int i = 0; i < 8; i++) b[i] = (__bf16)xa[i];
    if (C == 32){
      v8f xb = load8f(xr + 16 + 8 * half) * sc1 + sh1;
#pragma unroll
      for (int i = 0; i < 8; i++) b[8 + i] = (__bf16)xb[i];
    } else {
#pragma unroll
      for (int i = 0; i < 8; i++) b[8 + i] = (__bf16)0.f;
    }
    v8f cs0 = {}, cs1 = {}, cd0 = {}, cd1 = {}, cg0 = {}, cg1 = {};
    cs0 = wmma_bf16(aS0, b, cs0); cs1 = wmma_bf16(aS1, b, cs1);
    cd0 = wmma_bf16(aD0, b, cd0); cd1 = wmma_bf16(aD1, b, cd1);
    cg0 = wmma_bf16(aG0, b, cg0); cg1 = wmma_bf16(aG1, b, cg1);
    if (node < N){
      size_t base = (size_t)((unsigned)node) * 32u;
      store8f(ps + base + 8 * half, cs0);  store8f(ps + base + 16 + 8 * half, cs1);
      store8f(pd + base + 8 * half, cd0);  store8f(pd + base + 16 + 8 * half, cd1);
      store8f(xl + base + 8 * half, cg0);  store8f(xl + base + 16 + 8 * half, cg1);
    }
  }
}

// Edge MLP stage 1: h1 = lrelu( e@W1e + ps[src] + pd[dst] + b1_eff ), bn2 stats out.
template <int CE>
__global__ void k_edge_mm1_t(const unsigned short* __restrict__ eIn, int E,
                             const int* __restrict__ srcI, const int* __restrict__ dstI,
                             const float* __restrict__ ps, const float* __restrict__ pd,
                             const float* __restrict__ W1e, const float* __restrict__ beff,
                             unsigned short* __restrict__ h1,
                             float* __restrict__ osum, float* __restrict__ osq){
  const int lane = threadIdx.x & 31;
  const int half = lane >> 4;
  const int wave = (blockIdx.x * blockDim.x + threadIdx.x) >> 5;
  const int nw = (gridDim.x * blockDim.x) >> 5;

  v16bf a0 = load_wfragA(W1e, 0, lane, 32);
  v16bf a1 = load_wfragA(W1e, 16, lane, 32);
  v8f bf0 = load8f(beff + 8 * half);
  v8f bf1 = load8f(beff + 16 + 8 * half);

  v8f s0 = {}, q0 = {}, s1 = {}, q1 = {};
  const int ntiles = (E + 15) >> 4;
  for (int t = wave; t < ntiles; t += nw){
    int edge = t * 16 + (lane & 15);
    int ec = edge < E ? edge : E - 1;
    v16bf b = load_bfrag_bf16<CE>(eIn, ec, half);
    v8f c0 = {}, c1 = {};
    c0 = wmma_bf16(a0, b, c0);
    c1 = wmma_bf16(a1, b, c1);
    const bool valid = (edge < E);
    const int si = srcI[ec], di = dstI[ec];
    const float* psr = ps + (size_t)((unsigned)si) * 32u;
    const float* pdr = pd + (size_t)((unsigned)di) * 32u;
    v8f pa = load8f(psr + 8 * half);
    v8f pb = load8f(psr + 16 + 8 * half);
    v8f da = load8f(pdr + 8 * half);
    v8f db = load8f(pdr + 16 + 8 * half);
    v8f x0 = lrelu8(c0 + pa + da + bf0, 0.1f);
    v8f x1 = lrelu8(c1 + pb + db + bf1, 0.1f);
    if (valid){
      *(uint4*)(h1 + (size_t)((unsigned)edge) * 32u + 8 * half)      = pack_bf16x8(x0);
      *(uint4*)(h1 + (size_t)((unsigned)edge) * 32u + 16 + 8 * half) = pack_bf16x8(x1);
      s0 += x0; q0 += x0 * x0; s1 += x1; q1 += x1 * x1;
    }
  }
#pragma unroll
  for (int m = 1; m < 16; m <<= 1){
#pragma unroll
    for (int v = 0; v < 8; v++){
      s0[v] += __shfl_xor(s0[v], m); q0[v] += __shfl_xor(q0[v], m);
      s1[v] += __shfl_xor(s1[v], m); q1[v] += __shfl_xor(q1[v], m);
    }
  }
  if ((lane & 15) == 0){
#pragma unroll
    for (int v = 0; v < 8; v++){
      atomicAdd(&osum[8 * half + v],      s0[v]);
      atomicAdd(&osq [8 * half + v],      q0[v]);
      atomicAdd(&osum[16 + 8 * half + v], s1[v]);
      atomicAdd(&osq [16 + 8 * half + v], q1[v]);
    }
  }
}

// Edge MLP stage 2/3: out = act( in@Wf + beff ) with per-channel stats for the
// next BN. slope=0.1 => leaky relu; slope=1.0 => identity (final linear).
__global__ void k_edge_mm2(const unsigned short* __restrict__ eIn, int E,
                           const float* __restrict__ Wf, const float* __restrict__ beff,
                           float slope, unsigned short* __restrict__ out,
                           float* __restrict__ osum, float* __restrict__ osq){
  const int lane = threadIdx.x & 31;
  const int half = lane >> 4;
  const int wave = (blockIdx.x * blockDim.x + threadIdx.x) >> 5;
  const int nw = (gridDim.x * blockDim.x) >> 5;

  v16bf a0 = load_wfragA(Wf, 0, lane, 32);
  v16bf a1 = load_wfragA(Wf, 16, lane, 32);
  v8f bf0 = load8f(beff + 8 * half);
  v8f bf1 = load8f(beff + 16 + 8 * half);

  v8f s0 = {}, q0 = {}, s1 = {}, q1 = {};
  const int ntiles = (E + 15) >> 4;
  for (int t = wave; t < ntiles; t += nw){
    int edge = t * 16 + (lane & 15);
    int ec = edge < E ? edge : E - 1;
    v16bf b = load_bfrag_bf16<32>(eIn, ec, half);
    v8f c0 = {}, c1 = {};
    c0 = wmma_bf16(a0, b, c0);
    c1 = wmma_bf16(a1, b, c1);
    const bool valid = (edge < E);
    v8f x0 = lrelu8(c0 + bf0, slope);
    v8f x1 = lrelu8(c1 + bf1, slope);
    if (valid){
      *(uint4*)(out + (size_t)((unsigned)edge) * 32u + 8 * half)      = pack_bf16x8(x0);
      *(uint4*)(out + (size_t)((unsigned)edge) * 32u + 16 + 8 * half) = pack_bf16x8(x1);
      s0 += x0; q0 += x0 * x0; s1 += x1; q1 += x1 * x1;
    }
  }
#pragma unroll
  for (int m = 1; m < 16; m <<= 1){
#pragma unroll
    for (int v = 0; v < 8; v++){
      s0[v] += __shfl_xor(s0[v], m); q0[v] += __shfl_xor(q0[v], m);
      s1[v] += __shfl_xor(s1[v], m); q1[v] += __shfl_xor(q1[v], m);
    }
  }
  if ((lane & 15) == 0){
#pragma unroll
    for (int v = 0; v < 8; v++){
      atomicAdd(&osum[8 * half + v],      s0[v]);
      atomicAdd(&osq [8 * half + v],      q0[v]);
      atomicAdd(&osum[16 + 8 * half + v], s1[v]);
      atomicAdd(&osq [16 + 8 * half + v], q1[v]);
    }
  }
}

// ------------------------------ GAT kernels --------------------------------

__global__ void k_att(const float* __restrict__ xl, const float* __restrict__ attS,
                      const float* __restrict__ attD, int N,
                      float* __restrict__ asrc, float* __restrict__ adst){
  int n = blockIdx.x * blockDim.x + threadIdx.x;
  if (n >= N) return;
  const float* r = xl + (size_t)((unsigned)n) * 32u;
  float a1 = 0.f, a2 = 0.f;
  for (int c = 0; c < 32; c++){ a1 += r[c] * attS[c]; a2 += r[c] * attD[c]; }
  asrc[n] = a1; adst[n] = a2;
}

__global__ void k_gat_alpha(const int* __restrict__ src, const int* __restrict__ dst, int E, int N,
                            const float* __restrict__ as, const float* __restrict__ ad,
                            float* __restrict__ alpha, float* __restrict__ amax){
  int j = blockIdx.x * blockDim.x + threadIdx.x;
  if (j >= E + N) return;
  int r, c;
  if (j < E){ r = src[j]; c = dst[j]; } else { r = c = j - E; }
  float a = lrelu(as[r] + ad[c], 0.2f);
  alpha[j] = a;
  atomicMaxF(&amax[c], a);
}

__global__ void k_gat_soft(const int* __restrict__ src, const int* __restrict__ dst, int E, int N,
                           float* __restrict__ alpha, const float* __restrict__ amax,
                           float* __restrict__ den){
  int j = blockIdx.x * blockDim.x + threadIdx.x;
  if (j >= E + N) return;
  int c = (j < E) ? dst[j] : j - E;
  float ex = __expf(alpha[j] - amax[c]);
  alpha[j] = ex;
  atomicAdd(&den[c], ex);
}

__global__ void k_gat_agg(const int* __restrict__ src, const int* __restrict__ dst, int E, int N,
                          const float* __restrict__ ex, const float* __restrict__ den,
                          const float* __restrict__ xl, float* __restrict__ xout){
  int idx = blockIdx.x * blockDim.x + threadIdx.x;
  int tot = (E + N) * 8;
  if (idx >= tot) return;
  int j = idx >> 3; int q = idx & 7;
  int r, c;
  if (j < E){ r = src[j]; c = dst[j]; } else { r = c = j - E; }
  float w = ex[j] / (den[c] + 1e-16f);
  v4f v = *(const v4f*)(xl + (size_t)((unsigned)r) * 32u + q * 4);
  float* o = xout + (size_t)((unsigned)c) * 32u + q * 4;
  atomicAdd(o + 0, w * v[0]); atomicAdd(o + 1, w * v[1]);
  atomicAdd(o + 2, w * v[2]); atomicAdd(o + 3, w * v[3]);
}

__global__ void k_gat_out(const float* __restrict__ xout, const float* __restrict__ bias,
                          int N, float* __restrict__ xnew){
  int i = blockIdx.x * blockDim.x + threadIdx.x;
  if (i < N * 32) xnew[i] = lrelu(xout[i] + bias[i & 31], 0.1f);
}

// ------------------------------ heads --------------------------------------

__global__ void k_head_node(const float* __restrict__ x, const float* __restrict__ w,
                            const float* __restrict__ b, int N, float* __restrict__ out){
  int n = blockIdx.x * blockDim.x + threadIdx.x;
  if (n >= N) return;
  const float* r = x + (size_t)((unsigned)n) * 32u;
  float o0 = b[0], o1 = b[1];
  for (int c = 0; c < 32; c++){ float v = r[c]; o0 += v * w[c * 2]; o1 += v * w[c * 2 + 1]; }
  out[n * 2] = o0; out[n * 2 + 1] = o1;
}

__global__ void k_head_edge(const unsigned short* __restrict__ e, const float* __restrict__ w,
                            const float* __restrict__ b, int E, float* __restrict__ out){
  int j = blockIdx.x * blockDim.x + threadIdx.x;
  if (j >= E) return;
  const unsigned short* r = e + (size_t)((unsigned)j) * 32u;
  float o0 = b[0], o1 = b[1];
  for (int c = 0; c < 32; c++){ float v = bf2f(r[c]); o0 += v * w[c * 2]; o1 += v * w[c * 2 + 1]; }
  out[j * 2] = o0; out[j * 2 + 1] = o1;
}

// ------------------------------ launcher -----------------------------------
// Input leaf order assumption: top-level dict insertion order, params pytree
// flattened with dict keys sorted (JAX convention):
//   0 node_features, 1 edge_indices, 2 edge_features, 3 xbatch,
//   4 edge_b, 5 edge_w,
//   per layer l (6+18*l): bn_node_b, bn_node_g, b1, b2, b3, bn1_b, bn1_g,
//     bn2_b, bn2_g, bn3_b, bn3_g, w1, w2, w3, att_dst, att_src, bias, w(gat),
//   60 node_b, 61 node_w.

extern "C" void kernel_launch(void* const* d_in, const int* in_sizes, int n_in,
                              void* d_out, int out_size, void* d_ws, size_t ws_size,
                              hipStream_t stream){
  (void)n_in; (void)out_size; (void)ws_size;
  const int N = in_sizes[0] / 16;
  const int E = in_sizes[1] / 2;
  const float* x0  = (const float*)d_in[0];
  const int*   ei  = (const int*)d_in[1];
  const int*   srcI = ei;
  const int*   dstI = ei + E;
  const float* e0f = (const float*)d_in[2];
  const float* edge_b = (const float*)d_in[4];
  const float* edge_w = (const float*)d_in[5];
  const float* node_b = (const float*)d_in[60];
  const float* node_w = (const float*)d_in[61];
  auto LP = [&](int l, int o)->const float*{ return (const float*)d_in[6 + 18 * l + o]; };

  // workspace carve-out
  char* ws = (char*)d_ws;
  size_t off = 0;
  auto alloc = [&](size_t bytes)->char*{
    off = (off + 255) & ~(size_t)255;
    char* p = ws + off; off += bytes; return p;
  };
  unsigned short* eA  = (unsigned short*)alloc((size_t)E * 32 * 2);
  unsigned short* eB  = (unsigned short*)alloc((size_t)E * 32 * 2);
  unsigned short* e0b = (unsigned short*)alloc((size_t)E * 16 * 2);
  float* xX   = (float*)alloc((size_t)N * 32 * 4);
  float* xY   = (float*)alloc((size_t)N * 32 * 4);
  float* ps   = (float*)alloc((size_t)N * 32 * 4);
  float* pd   = (float*)alloc((size_t)N * 32 * 4);
  float* xl   = (float*)alloc((size_t)N * 32 * 4);
  float* asrc = (float*)alloc((size_t)N * 4);
  float* adst = (float*)alloc((size_t)N * 4);
  float* alpha= (float*)alloc((size_t)(E + N) * 4);
  float* amax = (float*)alloc((size_t)N * 4);
  float* den  = (float*)alloc((size_t)N * 4);
  float* xout = (float*)alloc((size_t)N * 32 * 4);
  float* small= (float*)alloc(320 * 4);
  float* s1e  = (float*)alloc(64 * 4); float* q1e = s1e + 32;
  float* scN  = (float*)alloc(64 * 4); float* shN = scN + 32;
  float* W1s  = (float*)alloc(1024 * 4);
  float* W1d  = (float*)alloc(1024 * 4);
  float* W1e  = (float*)alloc(1024 * 4);
  float* b1eff= (float*)alloc(32 * 4);
  float* W2f  = (float*)alloc(1024 * 4); float* b2eff = (float*)alloc(32 * 4);
  float* W3f  = (float*)alloc(1024 * 4); float* b3eff = (float*)alloc(32 * 4);
  int*   degs = (int*)alloc((size_t)N * 8); int* degd = degs + N;

  float* nsum = small;       float* nsq = small + 32;
  float* s1s  = small + 64;  float* q1s = small + 96;
  float* s1d  = small + 128; float* q1d = small + 160;
  float* s2   = small + 192; float* q2  = small + 224;
  float* s3   = small + 256; float* q3  = small + 288;

  dim3 B(256);
  auto NB = [&](long n)->dim3{ return dim3((unsigned)((n + 255) / 256)); };

  // --- prologue: degrees + layer-0 edge-feature stats + bf16 conversion
  k_zero<<<NB(2 * (long)N), B, 0, stream>>>((unsigned int*)degs, 2 * N);
  k_zero<<<1, 64, 0, stream>>>((unsigned int*)s1e, 64);
  k_deg<<<NB(E), B, 0, stream>>>(srcI, dstI, E, degs, degd);
  k_colstats<<<1024, B, 0, stream>>>(e0f, E * 16, 16, s1e, q1e);
  k_cvt_bf16<<<NB((long)E * 16), B, 0, stream>>>(e0f, e0b, E * 16);

  const float* xin = x0; int C = 16;
  unsigned short* eIn = e0b; int Ce = 16;
  unsigned short* h1p[3] = {eA, eB, eA};
  unsigned short* h2p[3] = {eB, eA, eB};
  unsigned short* eop[3] = {eA, eB, eA};
  float* xop[3] = {xX, xY, xX};

  for (int l = 0; l < 3; l++){
    k_layer_init<<<NB((long)N * 32), B, 0, stream>>>(small, amax, den, xout, N);
    // node BN stats + affine
    k_colstats<<<1024, B, 0, stream>>>(xin, N * C, C, nsum, nsq);
    k_fold_node<<<1, 64, 0, stream>>>(nsum, nsq, LP(l, 1), LP(l, 0), N, C, scN, shN);
    // bn1 x-segment stats via degree weighting; e-segment stats already in s1e/q1e
    k_bn1x_stats<<<1024, B, 0, stream>>>(xin, N, C, (C == 16 ? 4 : 5), scN, shN,
                                         degs, degd, s1s, q1s, s1d, q1d);
    k_fold1<<<1, 128, 0, stream>>>(LP(l, 11), LP(l, 6), LP(l, 5), LP(l, 2),
                                   s1s, q1s, s1d, q1d, s1e, q1e,
                                   C, Ce, E, W1s, W1d, W1e, b1eff);
    // per-node WMMA: ps, pd (edge-MLP partials) and xl (GAT transform)
    if (C == 16)
      k_node_mm_t<16><<<512, B, 0, stream>>>(xin, N, scN, shN, W1s, W1d, LP(l, 17), ps, pd, xl);
    else
      k_node_mm_t<32><<<512, B, 0, stream>>>(xin, N, scN, shN, W1s, W1d, LP(l, 17), ps, pd, xl);
    k_att<<<NB(N), B, 0, stream>>>(xl, LP(l, 15), LP(l, 14), N, asrc, adst);
    // edge MLP (WMMA, BN folded), stats cascaded between stages
    if (Ce == 16)
      k_edge_mm1_t<16><<<2048, B, 0, stream>>>(eIn, E, srcI, dstI, ps, pd, W1e, b1eff, h1p[l], s2, q2);
    else
      k_edge_mm1_t<32><<<2048, B, 0, stream>>>(eIn, E, srcI, dstI, ps, pd, W1e, b1eff, h1p[l], s2, q2);
    k_fold32<<<1, 128, 0, stream>>>(LP(l, 12), LP(l, 8), LP(l, 7), LP(l, 3), s2, q2, E, W2f, b2eff);
    k_edge_mm2<<<2048, B, 0, stream>>>(h1p[l], E, W2f, b2eff, 0.1f, h2p[l], s3, q3);
    k_fold32<<<1, 128, 0, stream>>>(LP(l, 13), LP(l, 10), LP(l, 9), LP(l, 4), s3, q3, E, W3f, b3eff);
    k_edge_mm2<<<2048, B, 0, stream>>>(h2p[l], E, W3f, b3eff, 1.0f, eop[l], s1e, q1e);
    // GAT softmax + aggregation
    k_gat_alpha<<<NB((long)E + N), B, 0, stream>>>(srcI, dstI, E, N, asrc, adst, alpha, amax);
    k_gat_soft <<<NB((long)E + N), B, 0, stream>>>(srcI, dstI, E, N, alpha, amax, den);
    k_gat_agg  <<<NB(((long)E + N) * 8), B, 0, stream>>>(srcI, dstI, E, N, alpha, den, xl, xout);
    k_gat_out  <<<NB((long)N * 32), B, 0, stream>>>(xout, LP(l, 16), N, xop[l]);
    xin = xop[l]; C = 32; eIn = eop[l]; Ce = 32;
  }

  float* xpred = (float*)d_out;
  float* epred = (float*)d_out + 2 * N;
  k_head_node<<<NB(N), B, 0, stream>>>(xin, node_w, node_b, N, xpred);
  k_head_edge<<<NB(E), B, 0, stream>>>(eIn, edge_w, edge_b, E, epred);
}
